// GravTransformer_70557722739199
// MI455X (gfx1250) — matrix operand order repaired and checked by reading, not compiled, not run on hardware
//
#include <hip/hip_runtime.h>
#include <hip/hip_bf16.h>

// ---------------------------------------------------------------------------
// GravTransformer on MI455X (gfx1250).
// - All GEMMs: v_wmma_f32_16x16x32_bf16, fp32 accumulate.
// - Weights pre-packed ONCE to bf16 in exact WMMA B-fragment order
//   (tile-major, lane&15 = N index, vector element = K index).
// - Each GEMM wave owns 1 M-tile x 4 N-tiles: the expensive fp32 A fragment
//   is loaded+converted once per k-step and feeds 4 WMMAs against 4 cheap
//   packed-bf16 B fragments (192 B/lane per 4 WMMAs vs 288 B before).
// - Attention fused per (batch, head, 16-query tile): scores + softmax live
//   in a 32KB LDS tile, both QK^T and P*V on the WMMA pipe.
// - Problem is memory bound (~300MB HBM traffic vs ~70 GFLOP): activations
//   stay fp32 in HBM (reference-identical bytes), bf16 only inside the MACs.
// ---------------------------------------------------------------------------

typedef __attribute__((ext_vector_type(16))) __bf16 v16bf;
typedef __attribute__((ext_vector_type(8)))  float  v8f;

#define D_MODEL 256
#define SEQ     512
#define BATCH   16
#define HEADS   8
#define HEAD_D  32
#define LAYERS  4
#define N_TOK   (BATCH * SEQ)      // 8192
#define N_EDGE  65536
#define N_PRED  32768
#define DFF     512
#define K_CLS   16

// --- fragment loaders (layouts per CDNA5 ISA 7.12.2) -----------------------

// A fragment: 16x32 tile from row-major fp32.
// lane&15 = M row; lanes<16 hold K {0..7,16..23}, lanes>=16 hold K {8..15,24..31}.
__device__ __forceinline__ v16bf load_a_frag(const float* __restrict__ rowptr,
                                             int kb /* (lane>>4)*8 */) {
  v16bf a;
#pragma unroll
  for (int i = 0; i < 4; ++i) {
    a[2 * i]         = (__bf16)rowptr[kb + 2 * i];
    a[2 * i + 1]     = (__bf16)rowptr[kb + 2 * i + 1];
    a[8 + 2 * i]     = (__bf16)rowptr[16 + kb + 2 * i];
    a[8 + 2 * i + 1] = (__bf16)rowptr[16 + kb + 2 * i + 1];
  }
  return a;
}

// B fragment (K=32 x N=16) built on the fly from fp32 W[N,K] (B = W^T):
// lane&15 = N index; lanes<16 hold K {0..7,16..23}, lanes>=16 K {8..15,24..31}. (Used in attention.)
__device__ __forceinline__ v16bf load_b_frag_wt(const float* __restrict__ W,
                                                long ldw, int n0, int k) {
  v16bf b;
  const float* wp = W + (long)(n0 + (k & 15)) * ldw + ((k >> 4) << 3);
#pragma unroll
  for (int j = 0; j < 16; ++j) b[j] = (__bf16)wp[((j >> 3) << 4) + (j & 7)];
  return b;
}

// B fragment from row-major fp32 B[K,N]: lane&15 = N index, elements = K in table order.
__device__ __forceinline__ v16bf load_b_frag_rm(const float* __restrict__ Bm,
                                                long ldb, int k, int n0) {
  v16bf b;
  const int l = k & 31;
  const float* rp = Bm + (long)(k - l + ((l >> 4) << 3)) * ldb + n0 + (l & 15);
#pragma unroll
  for (int j = 0; j < 16; ++j) b[j] = (__bf16)rp[(long)(((j >> 3) << 4) + (j & 7)) * ldb];
  return b;
}

// --- weight pre-pack: fp32 W[N,K] -> bf16 fragments, tile-major ------------
// dst element [(nt*(K/32)+kt)*512 + lane*16 + j] = W[(nt*16+(lane&15))*K + kt*32 + ((j>>3)<<4)+((lane>>4)<<3)+(j&7)]
__global__ __launch_bounds__(512) void pack_w_kernel(
    const float* __restrict__ W, __bf16* __restrict__ dst, int N, int K) {
  const int tile = blockIdx.x;           // (N/16)*(K/32) tiles
  const int ktiles = K >> 5;
  const int nt = tile / ktiles, kt = tile % ktiles;
  const int t = threadIdx.x;             // 0..511
  const int lane = t >> 4, j = t & 15;
  dst[(long)tile * 512 + t] =
      (__bf16)W[(long)(nt * 16 + (lane & 15)) * K + kt * 32 + ((j >> 3) << 4) + ((lane >> 4) << 3) + (j & 7)];
}

// --- generic WMMA GEMM ------------------------------------------------------
// C[M,N] = op(A[M,K] (optionally row-gathered) @ W^T + bias) (+ residual)
// Wsw is the pre-packed bf16 fragment stream. 4 waves/block; each wave owns
// 1 M-tile x 4 N-tiles, so the fp32 A fragment is converted once and feeds
// 4 WMMAs. A block spans 256 output columns.
__global__ __launch_bounds__(128) void wmma_gemm_kernel(
    const float* __restrict__ A, const int* __restrict__ gidx,
    const __bf16* __restrict__ Wsw, const float* __restrict__ bias,
    const float* __restrict__ residual, float* __restrict__ C,
    int M, int N, int Kd, int relu) {
  const int lane = threadIdx.x & 31;
  const int wave = threadIdx.x >> 5;
  const int m0 = blockIdx.x * 16;
  const int ncol0 = blockIdx.y * 256 + wave * 64;  // this wave's 4 N-tiles
  if (ncol0 >= N) return;  // uniform per wave

  const int ktiles = Kd >> 5;
  const v16bf* bw[4];
#pragma unroll
  for (int nt = 0; nt < 4; ++nt)
    bw[nt] = (const v16bf*)Wsw + ((long)((ncol0 >> 4) + nt) * ktiles) * 32 + lane;

  const int kb = (lane >> 4) << 3;
  const int mr = m0 + (lane & 15);
  const long arow = gidx ? (long)gidx[mr] : (long)mr;
  const float* arp = A + arow * (long)Kd;

  v8f c[4] = {};
  for (int k0 = 0; k0 < Kd; k0 += 32) {
    if (k0 + 32 < Kd)  // gfx1250 global_prefetch of next fp32 A step
      __builtin_prefetch(arp + k0 + 32, 0, 3);
    const v16bf a = load_a_frag(arp + k0, kb);  // converted once, used 4x
    const int ko = (k0 >> 5) * 32;
    const v16bf b0 = bw[0][ko];
    const v16bf b1 = bw[1][ko];
    const v16bf b2 = bw[2][ko];
    const v16bf b3 = bw[3][ko];
    c[0] = __builtin_amdgcn_wmma_f32_16x16x32_bf16(false, a, false, b0, (short)0, c[0], false, false);
    c[1] = __builtin_amdgcn_wmma_f32_16x16x32_bf16(false, a, false, b1, (short)0, c[1], false, false);
    c[2] = __builtin_amdgcn_wmma_f32_16x16x32_bf16(false, a, false, b2, (short)0, c[2], false, false);
    c[3] = __builtin_amdgcn_wmma_f32_16x16x32_bf16(false, a, false, b3, (short)0, c[3], false, false);
  }

  const int mbase = m0 + ((lane >> 4) << 3);
#pragma unroll
  for (int nt = 0; nt < 4; ++nt) {
    const int nc = ncol0 + nt * 16 + (lane & 15);
    const float bv = bias ? bias[nc] : 0.0f;
#pragma unroll
    for (int r = 0; r < 8; ++r) {
      const long mc = mbase + r;
      float v = c[nt][r] + bv;
      if (residual) v += residual[mc * N + nc];
      if (relu) v = fmaxf(v, 0.0f);
      C[mc * N + nc] = v;
    }
  }
}

// --- fused attention: one wave per (batch, head, 16-query tile) ------------
// qkv layout [N_TOK, 768]: cols 0..255 Q, 256..511 K, 512..767 V.
__global__ __launch_bounds__(32) void attn_kernel(
    const float* __restrict__ qkv, const unsigned char* __restrict__ mask,
    float* __restrict__ out) {
  __shared__ float sc[16 * SEQ];  // 32 KB score/prob tile
  const int lane = threadIdx.x;
  const int q0 = blockIdx.x * 16;
  const int h  = blockIdx.y;
  const int bt = blockIdx.z;

  const float* base = qkv + (long)bt * SEQ * 768;
  const float* qp = base + h * HEAD_D;
  const float* kp = base + D_MODEL + h * HEAD_D;
  const float* vp = base + 2 * D_MODEL + h * HEAD_D;
  const int kb = (lane >> 4) << 3;
  const float scale = 0.1767766952966369f;  // 1/sqrt(32)

  // Q fragment: 16 queries x 32 dims (single K-step)
  const v16bf aq = load_a_frag(qp + (long)(q0 + (lane & 15)) * 768, kb);

  // scores -> LDS
  for (int kt = 0; kt < SEQ / 16; ++kt) {
    v16bf bk = load_b_frag_wt(kp, 768, kt * 16, lane);  // B = K^T, K-dim = head dim
    v8f c = {};
    c = __builtin_amdgcn_wmma_f32_16x16x32_bf16(false, aq, false, bk,
                                                (short)0, c, false, false);
    const int col = kt * 16 + (lane & 15);
    const float mb = mask[bt * SEQ + col] ? -1e9f : 0.0f;
    const int rbase = (lane >> 4) << 3;
#pragma unroll
    for (int r = 0; r < 8; ++r) sc[(rbase + r) * SEQ + col] = c[r] * scale + mb;
  }
  __syncthreads();

  // row softmax (lane l handles query row l)
  if (lane < 16) {
    float* row = sc + lane * SEQ;
    float mx = -1e30f;
    for (int j = 0; j < SEQ; ++j) mx = fmaxf(mx, row[j]);
    float s = 0.0f;
    for (int j = 0; j < SEQ; ++j) { float e = __expf(row[j] - mx); row[j] = e; s += e; }
    const float inv = 1.0f / s;
    for (int j = 0; j < SEQ; ++j) row[j] *= inv;
  }
  __syncthreads();

  // O(16x32) = P(16x512) @ V(512x32)
  for (int nt = 0; nt < 2; ++nt) {
    v8f c = {};
    for (int kt = 0; kt < SEQ / 32; ++kt) {
      v16bf ap = load_a_frag(sc + (lane & 15) * SEQ + kt * 32, kb);
      v16bf bv = load_b_frag_rm(vp, 768, kt * 32 + lane, nt * 16);
      c = __builtin_amdgcn_wmma_f32_16x16x32_bf16(false, ap, false, bv,
                                                  (short)0, c, false, false);
    }
    const int col = nt * 16 + (lane & 15);
    const int rbase = (lane >> 4) << 3;
#pragma unroll
    for (int r = 0; r < 8; ++r) {
      const long q = q0 + rbase + r;
      out[((long)bt * SEQ + q) * D_MODEL + h * HEAD_D + col] = c[r];
    }
  }
}

// --- x = node_embedding + tiled pos_table ----------------------------------
__global__ void init_x_kernel(const float* __restrict__ node,
                              const float* __restrict__ pos,
                              float* __restrict__ x) {
  const long i = (long)blockIdx.x * 256 + threadIdx.x;
  const int row = (int)(i >> 8), d = (int)(i & 255);
  x[i] = node[i] + pos[((row % SEQ) << 8) + d];
}

// --- scatter-add of messages into x ---------------------------------------
__global__ void scatter_add_kernel(const float* __restrict__ msgs,
                                   const int* __restrict__ inc0,
                                   float* __restrict__ x) {
  const int e = blockIdx.x, d = threadIdx.x;
  atomicAdd(&x[(long)inc0[e] * D_MODEL + d], msgs[(long)e * D_MODEL + d]);
}

// --- LayerNorm over D=256, one block per row -------------------------------
__global__ __launch_bounds__(256) void layernorm_kernel(
    const float* __restrict__ in, const float* __restrict__ g,
    const float* __restrict__ b, float* __restrict__ out) {
  __shared__ float red[256];
  const int row = blockIdx.x, t = threadIdx.x;
  const float v = in[(long)row * 256 + t];
  red[t] = v; __syncthreads();
  for (int s = 128; s > 0; s >>= 1) { if (t < s) red[t] += red[t + s]; __syncthreads(); }
  const float mu = red[0] * (1.0f / 256.0f);
  __syncthreads();
  const float d = v - mu;
  red[t] = d * d; __syncthreads();
  for (int s = 128; s > 0; s >>= 1) { if (t < s) red[t] += red[t + s]; __syncthreads(); }
  const float var = red[0] * (1.0f / 256.0f);
  out[(long)row * 256 + t] = d * rsqrtf(var + 1e-5f) * g[t] + b[t];
}

// --- edge heads: rep = x[a]*x[b]; scalar head + optional 16-way type head --
__global__ __launch_bounds__(256) void edge_pred_kernel(
    const float* __restrict__ x, const int* __restrict__ edges,
    const float* __restrict__ pe_w, const float* __restrict__ pe_b,
    const float* __restrict__ pt_w, const float* __restrict__ pt_b,
    float* __restrict__ pred_s, float* __restrict__ pred_t) {
  __shared__ float rep[256];
  __shared__ float red[256];
  const int e = blockIdx.x, t = threadIdx.x;
  const int ia = edges[2 * e], ib = edges[2 * e + 1];
  rep[t] = x[(long)ia * 256 + t] * x[(long)ib * 256 + t];
  __syncthreads();
  red[t] = rep[t] * pe_w[t]; __syncthreads();
  for (int s = 128; s > 0; s >>= 1) { if (t < s) red[t] += red[t + s]; __syncthreads(); }
  if (t == 0) pred_s[e] = red[0] + pe_b[0];
  if (pred_t) {
    for (int kk = 0; kk < K_CLS; ++kk) {
      __syncthreads();
      red[t] = rep[t] * pt_w[kk * 256 + t]; __syncthreads();
      for (int s = 128; s > 0; s >>= 1) { if (t < s) red[t] += red[t + s]; __syncthreads(); }
      if (t == 0) pred_t[(long)e * K_CLS + kk] = red[0] + pt_b[kk];
    }
  }
}

// ---------------------------------------------------------------------------
extern "C" void kernel_launch(void* const* d_in, const int* in_sizes, int n_in,
                              void* d_out, int out_size, void* d_ws, size_t ws_size,
                              hipStream_t stream) {
  const float* node       = (const float*)d_in[0];
  const float* edge       = (const float*)d_in[1];
  const int*   incidence  = (const int*)d_in[2];
  const unsigned char* mask = (const unsigned char*)d_in[3];
  const int*   edges_neg  = (const int*)d_in[4];
  const int*   edges_pos  = (const int*)d_in[5];
  const float* msg_w1     = (const float*)d_in[6];
  const float* msg_w2     = (const float*)d_in[7];
  const float* msg_b      = (const float*)d_in[8];
  const float* pos_table  = (const float*)d_in[9];
  const float* in_proj_w  = (const float*)d_in[10];
  const float* in_proj_b  = (const float*)d_in[11];
  const float* out_w      = (const float*)d_in[12];
  const float* out_b      = (const float*)d_in[13];
  const float* ln1_g      = (const float*)d_in[14];
  const float* ln1_b      = (const float*)d_in[15];
  const float* ff_w1      = (const float*)d_in[16];
  const float* ff_b1      = (const float*)d_in[17];
  const float* ff_w2      = (const float*)d_in[18];
  const float* ff_b2      = (const float*)d_in[19];
  const float* ln2_g      = (const float*)d_in[20];
  const float* ln2_b      = (const float*)d_in[21];
  const float* pe_w       = (const float*)d_in[22];
  const float* pe_b       = (const float*)d_in[23];
  const float* pt_w       = (const float*)d_in[24];
  const float* pt_b       = (const float*)d_in[25];

  // workspace layout (fp32 region + bf16 packed-weight region)
  float* ws   = (float*)d_ws;
  float* x    = ws;                                  // 8192*256
  float* msgs = x + (long)N_TOK * D_MODEL;           // 65536*256 (dead after scatter)
  float* qkv  = msgs;                                // 8192*768   (reuse)
  float* attn = qkv + (long)N_TOK * 3 * D_MODEL;     // 8192*256
  float* tmp  = attn + (long)N_TOK * D_MODEL;        // 8192*256
  float* ffh  = tmp + (long)N_TOK * D_MODEL;         // 8192*512

  __bf16* wpack = (__bf16*)(msgs + (long)N_EDGE * D_MODEL);
  const long SZ_DD  = (long)D_MODEL * D_MODEL;       // 65536
  const long SZ_QKV = (long)3 * D_MODEL * D_MODEL;   // 196608
  const long SZ_FF  = (long)DFF * D_MODEL;           // 131072
  __bf16* pw_msg2 = wpack;
  __bf16* pw_msg1 = pw_msg2 + SZ_DD;
  __bf16* pw_lay  = pw_msg1 + SZ_DD;                 // per layer: qkv,out,ff1,ff2
  const long SZ_LAY = SZ_QKV + SZ_DD + 2 * SZ_FF;

  const dim3 blk(128);

  // ---- pack all weights to bf16 WMMA-fragment order (once) ----
  pack_w_kernel<<<(D_MODEL / 16) * (D_MODEL / 32), 512, 0, stream>>>(msg_w2, pw_msg2, D_MODEL, D_MODEL);
  pack_w_kernel<<<(D_MODEL / 16) * (D_MODEL / 32), 512, 0, stream>>>(msg_w1, pw_msg1, D_MODEL, D_MODEL);
  for (int i = 0; i < LAYERS; ++i) {
    __bf16* pl = pw_lay + i * SZ_LAY;
    pack_w_kernel<<<(3 * D_MODEL / 16) * (D_MODEL / 32), 512, 0, stream>>>(
        in_proj_w + (long)i * SZ_QKV, pl, 3 * D_MODEL, D_MODEL);
    pack_w_kernel<<<(D_MODEL / 16) * (D_MODEL / 32), 512, 0, stream>>>(
        out_w + (long)i * SZ_DD, pl + SZ_QKV, D_MODEL, D_MODEL);
    pack_w_kernel<<<(DFF / 16) * (D_MODEL / 32), 512, 0, stream>>>(
        ff_w1 + (long)i * SZ_FF, pl + SZ_QKV + SZ_DD, DFF, D_MODEL);
    pack_w_kernel<<<(D_MODEL / 16) * (DFF / 32), 512, 0, stream>>>(
        ff_w2 + (long)i * SZ_FF, pl + SZ_QKV + SZ_DD + SZ_FF, D_MODEL, DFF);
  }

  // ---- message passing ----
  // msgs = edge_embedding @ msg_w2^T + msg_b
  wmma_gemm_kernel<<<dim3(N_EDGE / 16, D_MODEL / 256), blk, 0, stream>>>(
      edge, nullptr, pw_msg2, msg_b, nullptr, msgs, N_EDGE, D_MODEL, D_MODEL, 0);
  // msgs += node_embedding[incidence[1]] @ msg_w1^T
  wmma_gemm_kernel<<<dim3(N_EDGE / 16, D_MODEL / 256), blk, 0, stream>>>(
      node, incidence + N_EDGE, pw_msg1, nullptr, msgs, msgs, N_EDGE, D_MODEL, D_MODEL, 0);
  // x = node + tiled pos_table, then scatter-add msgs at incidence[0]
  init_x_kernel<<<N_TOK, 256, 0, stream>>>(node, pos_table, x);
  scatter_add_kernel<<<N_EDGE, 256, 0, stream>>>(msgs, incidence, x);

  // ---- transformer layers ----
  for (int i = 0; i < LAYERS; ++i) {
    __bf16* pl = pw_lay + i * SZ_LAY;
    const float* bqkv = in_proj_b + (long)i * 3 * D_MODEL;
    const float* ob = out_b + (long)i * D_MODEL;
    const float* f1b = ff_b1 + (long)i * DFF;
    const float* f2b = ff_b2 + (long)i * D_MODEL;

    wmma_gemm_kernel<<<dim3(N_TOK / 16, (3 * D_MODEL) / 256), blk, 0, stream>>>(
        x, nullptr, pl, bqkv, nullptr, qkv, N_TOK, 3 * D_MODEL, D_MODEL, 0);
    attn_kernel<<<dim3(SEQ / 16, HEADS, BATCH), 32, 0, stream>>>(qkv, mask, attn);
    wmma_gemm_kernel<<<dim3(N_TOK / 16, D_MODEL / 256), blk, 0, stream>>>(
        attn, nullptr, pl + SZ_QKV, ob, x, tmp, N_TOK, D_MODEL, D_MODEL, 0);
    layernorm_kernel<<<N_TOK, 256, 0, stream>>>(tmp, ln1_g + (long)i * D_MODEL,
                                                ln1_b + (long)i * D_MODEL, x);
    wmma_gemm_kernel<<<dim3(N_TOK / 16, DFF / 256), blk, 0, stream>>>(
        x, nullptr, pl + SZ_QKV + SZ_DD, f1b, nullptr, ffh, N_TOK, DFF, D_MODEL, 1);
    wmma_gemm_kernel<<<dim3(N_TOK / 16, D_MODEL / 256), blk, 0, stream>>>(
        ffh, nullptr, pl + SZ_QKV + SZ_DD + SZ_FF, f2b, x, tmp, N_TOK, D_MODEL, DFF, 0);
    layernorm_kernel<<<N_TOK, 256, 0, stream>>>(tmp, ln2_g + (long)i * D_MODEL,
                                                ln2_b + (long)i * D_MODEL, x);
  }

  // ---- edge heads: [pred_pos (NE)] [pred_neg (NE)] [pred_type (NE*16)] ----
  float* out = (float*)d_out;
  edge_pred_kernel<<<N_PRED, 256, 0, stream>>>(x, edges_pos, pe_w, pe_b, pt_w, pt_b,
                                               out, out + 2 * N_PRED);
  edge_pred_kernel<<<N_PRED, 256, 0, stream>>>(x, edges_neg, pe_w, pe_b, nullptr, nullptr,
                                               out + N_PRED, nullptr);
}
